// RelativePositionalAttention_88381837017268
// MI455X (gfx1250) — matrix operand order repaired and checked by reading, not compiled, not run on hardware
//
#include <hip/hip_runtime.h>

// ---------------------------------------------------------------------------
// Relative positional attention for MI455X (gfx1250), bf16 WMMA pipeline.
// - All GEMMs are NT (B stored [N,K]); V pre-transposed to keep B loads
//   contiguous (global_load_b128 pairs).
// - 32-bit in-kernel addressing.
// - K loop manually unrolled x2 with two named buffer sets: no register
//   ping-pong copies, loads for one half in flight while WMMAs run on the
//   other.
// - rel tables padded to 320 rows so B-fragment loads need no guards.
// ---------------------------------------------------------------------------

typedef __attribute__((ext_vector_type(16))) __bf16 v16bf;
typedef __attribute__((ext_vector_type(8)))  float  v8f;

#define SEQ    1024
#define BATCH  4
#define NHEAD  16
#define DK     64
#define DMODEL 1024
#define RELROWS 257
#define RELPAD  272   // P row length (17*16)
#define RELBUF  320   // rel table rows, padded to 5 full 64-wide N tiles
#define NTOK   (SEQ * BATCH)          // 4096

__device__ __forceinline__ int clamp_rel(int r) {
    r = (r < -128) ? -128 : r;
    r = (r >  128) ?  128 : r;
    return r + 128;
}

// ---- WMMA fragment loaders (contiguous k order, same for A and B, wave32) --------------
// A fragment: 16x32 bf16, row m = lane&15, element e -> k = (lane>>4)*16 + e.
__device__ __forceinline__ v16bf load_a_frag(const __bf16* __restrict__ A,
                                             int lda, int mbase, int kbase,
                                             int lane) {
    const int m  = mbase + (lane & 15);
    const int hi = lane >> 4;
    const int base = m * lda + kbase + hi * 16;
    v16bf r;
#pragma unroll
    for (int e = 0; e < 16; ++e) {
        r[e] = A[base + e];            // K run: hi*16 .. hi*16+15
    }
    return r;
}

// B fragment: 32x16 bf16, col n = lane&15, element e -> k = (lane>>4)*16 + e.
// Source layout [N,K] row-major: 16 contiguous elements, unconditional.
__device__ __forceinline__ v16bf load_b_frag(const __bf16* __restrict__ B,
                                             int ldb, int nbase, int kbase,
                                             int lane) {
    const int n  = nbase + (lane & 15);
    const int hi = lane >> 4;
    const int base = n * ldb + kbase + hi * 16;
    v16bf r;
#pragma unroll
    for (int e = 0; e < 16; ++e) r[e] = B[base + e];
    return r;
}

// ---- cast fp32 -> bf16 with zero tail --------------------------------------
__global__ void cast_bf16_kernel(const float* __restrict__ src,
                                 __bf16* __restrict__ dst,
                                 int nsrc, int ntot) {
    int i = blockIdx.x * blockDim.x + threadIdx.x;
    if (i < ntot) dst[i] = (i < nsrc) ? (__bf16)src[i] : (__bf16)0.0f;
}

// ---- V transpose: V[(s*4+b), n] -> Vt[(b*1024+n), s] -----------------------
__global__ __launch_bounds__(256) void transpose_v_kernel(
    const __bf16* __restrict__ V, __bf16* __restrict__ Vt) {
    __shared__ __bf16 tile[32][33];
    const int b  = blockIdx.z;
    const int s0 = blockIdx.x * 32;
    const int n0 = blockIdx.y * 32;
    const int tx = threadIdx.x;   // 0..31
    const int ty = threadIdx.y;   // 0..7
#pragma unroll
    for (int i = ty; i < 32; i += 8) {
        const int s = s0 + i;
        tile[i][tx] = V[((s * BATCH + b) << 10) + n0 + tx];
    }
    __syncthreads();
#pragma unroll
    for (int i = ty; i < 32; i += 8) {
        const int n = n0 + i;
        Vt[(((b << 10) + n) << 10) + s0 + tx] = tile[tx][i];
    }
}

// ---- generic batched bf16 GEMM (NT): C[z] = A[z](MxK) @ B[z](NxK)^T --------
// Block: 256 threads = 8 waves, 4x2 wave grid, wave tile 32x32, block 128x64.
// K loop unrolled x2 with two explicit buffer sets (no swap copies).
// Requires K == 64 or K % 64 == 0.
__global__ __launch_bounds__(256) void gemm_bf16_kernel(
    const __bf16* __restrict__ Ag, long sA, int lda,
    const __bf16* __restrict__ Bg, long sB, int ldb,
    float* __restrict__ Cf, __bf16* __restrict__ Cb, long sC, int ldc,
    int N, int K, const float* __restrict__ bias) {
    const int z = blockIdx.z;
    const __bf16* A = Ag + (long)z * sA;
    const __bf16* B = Bg + (long)z * sB;
    const long cbase = (long)z * sC;

    const int wave = threadIdx.x >> 5;
    const int lane = threadIdx.x & 31;
    const int wm = wave >> 1;          // 0..3
    const int wn = wave & 1;           // 0..1
    const int m0 = blockIdx.x * 128 + wm * 32;
    const int n0 = blockIdx.y * 64  + wn * 32;

    v8f acc00 = {}, acc01 = {}, acc10 = {}, acc11 = {};

    // buffer set A: preload k=0
    v16bf a0A = load_a_frag(A, lda, m0,      0, lane);
    v16bf a1A = load_a_frag(A, lda, m0 + 16, 0, lane);
    v16bf b0A = load_b_frag(B, ldb, n0,      0, lane);
    v16bf b1A = load_b_frag(B, ldb, n0 + 16, 0, lane);

    for (int k0 = 0; k0 < K; k0 += 64) {
        // prefetch buffer set B at k0+32 (always valid: K%64==0 -> k0+32<K)
        const int kB = k0 + 32;
        v16bf a0B = load_a_frag(A, lda, m0,      kB, lane);
        v16bf a1B = load_a_frag(A, lda, m0 + 16, kB, lane);
        v16bf b0B = load_b_frag(B, ldb, n0,      kB, lane);
        v16bf b1B = load_b_frag(B, ldb, n0 + 16, kB, lane);

        acc00 = __builtin_amdgcn_wmma_f32_16x16x32_bf16(false, a0A, false, b0A, (short)0, acc00, false, false);
        acc01 = __builtin_amdgcn_wmma_f32_16x16x32_bf16(false, a0A, false, b1A, (short)0, acc01, false, false);
        acc10 = __builtin_amdgcn_wmma_f32_16x16x32_bf16(false, a1A, false, b0A, (short)0, acc10, false, false);
        acc11 = __builtin_amdgcn_wmma_f32_16x16x32_bf16(false, a1A, false, b1A, (short)0, acc11, false, false);

        // prefetch buffer set A at k0+64 (clamped redundant load on last iter)
        const int kA = (k0 + 64 < K) ? (k0 + 64) : 0;
        a0A = load_a_frag(A, lda, m0,      kA, lane);
        a1A = load_a_frag(A, lda, m0 + 16, kA, lane);
        b0A = load_b_frag(B, ldb, n0,      kA, lane);
        b1A = load_b_frag(B, ldb, n0 + 16, kA, lane);

        acc00 = __builtin_amdgcn_wmma_f32_16x16x32_bf16(false, a0B, false, b0B, (short)0, acc00, false, false);
        acc01 = __builtin_amdgcn_wmma_f32_16x16x32_bf16(false, a0B, false, b1B, (short)0, acc01, false, false);
        acc10 = __builtin_amdgcn_wmma_f32_16x16x32_bf16(false, a1B, false, b0B, (short)0, acc10, false, false);
        acc11 = __builtin_amdgcn_wmma_f32_16x16x32_bf16(false, a1B, false, b1B, (short)0, acc11, false, false);
    }

    const int hi = lane >> 4;
    const int nn = lane & 15;
    const v8f* accs[2][2] = {{&acc00, &acc01}, {&acc10, &acc11}};
#pragma unroll
    for (int i = 0; i < 2; ++i)
#pragma unroll
        for (int j = 0; j < 2; ++j)
#pragma unroll
            for (int v = 0; v < 8; ++v) {
                const int m = m0 + i * 16 + v + 8 * hi;
                const int n = n0 + j * 16 + nn;
                if (n < N) {
                    float val = (*accs[i][j])[v] + (bias ? bias[n] : 0.0f);
                    const int idx = m * ldc + n;
                    if (Cb) Cb[cbase + idx] = (__bf16)val;
                    else    Cf[cbase + idx] = val;
                }
            }
}

// ---- fused scores + head-softmax ------------------------------------------
// grid (S/16, T/16, B), block 512 = 16 waves; wave w handles head h=w.
// scores[s,b,t,h] = (Q.K + P[s,b,h,idx(s,t)]) * scale ; softmax over h.
__global__ __launch_bounds__(512) void scores_softmax_kernel(
    const __bf16* __restrict__ Qb, const __bf16* __restrict__ Kb,
    const float* __restrict__ P, __bf16* __restrict__ attn) {
    const int s0 = blockIdx.x * 16;
    const int t0 = blockIdx.y * 16;
    const int b  = blockIdx.z;
    const int h    = threadIdx.x >> 5;
    const int lane = threadIdx.x & 31;

    __shared__ float sc[NHEAD * 16 * 16];   // [h][m][n], 16 KB

    const __bf16* Abase = Qb + b * DMODEL + h * DK;  // lda = BATCH*DMODEL
    const __bf16* Bbase = Kb + b * DMODEL + h * DK;

    // K = 64: two fully unrolled steps, both load groups issued up front.
    v16bf aA  = load_a_frag(Abase, BATCH * DMODEL, s0, 0,  lane);
    v16bf bbA = load_b_frag(Bbase, BATCH * DMODEL, t0, 0,  lane);
    v16bf aB  = load_a_frag(Abase, BATCH * DMODEL, s0, 32, lane);
    v16bf bbB = load_b_frag(Bbase, BATCH * DMODEL, t0, 32, lane);
    v8f acc = {};
    acc = __builtin_amdgcn_wmma_f32_16x16x32_bf16(false, aA, false, bbA, (short)0, acc, false, false);
    acc = __builtin_amdgcn_wmma_f32_16x16x32_bf16(false, aB, false, bbB, (short)0, acc, false, false);

    const float scale = 0.125f;  // 1/sqrt(64)
    const int hi = lane >> 4;
    const int n_ = lane & 15;
    const int t  = t0 + n_;
#pragma unroll
    for (int v = 0; v < 8; ++v) {
        const int m = v + 8 * hi;
        const int s = s0 + m;
        const int rel = clamp_rel(s - t);
        const float p = P[((s * BATCH + b) * NHEAD + h) * RELPAD + rel];
        sc[h * 256 + m * 16 + n_] = (acc[v] + p) * scale;
    }
    __syncthreads();

    if (threadIdx.x < 256) {
        const int m = threadIdx.x >> 4;
        const int n = threadIdx.x & 15;
        float vals[NHEAD];
        float mx = -3.4e38f;
#pragma unroll
        for (int h2 = 0; h2 < NHEAD; ++h2) {
            vals[h2] = sc[h2 * 256 + m * 16 + n];
            mx = fmaxf(mx, vals[h2]);
        }
        float sum = 0.0f;
#pragma unroll
        for (int h2 = 0; h2 < NHEAD; ++h2) {
            vals[h2] = __expf(vals[h2] - mx);
            sum += vals[h2];
        }
        const float inv = 1.0f / sum;
        const int s = s0 + m, tt = t0 + n;
#pragma unroll
        for (int h2 = 0; h2 < NHEAD; ++h2)
            attn[((b * NHEAD + h2) * SEQ + s) * SEQ + tt] =
                (__bf16)(vals[h2] * inv);
    }
}

// ---- out2[s,b,h,d] += sum_t attn[b,h,s,t] * rel_v[idx(s,t),d] -------------
// grid (S, B), block 128 = 4 waves; wave w covers d tile [16w, 16w+16).
// t loop unrolled x2; both halves' A loads + table gathers issue before the
// WMMA pair for latency hiding.
__global__ __launch_bounds__(128) void attn_relv_kernel(
    const __bf16* __restrict__ attn, const __bf16* __restrict__ relv,
    float* __restrict__ out) {
    const int s = blockIdx.x;
    const int b = blockIdx.y;
    const int w    = threadIdx.x >> 5;
    const int lane = threadIdx.x & 31;
    const int d0 = w * 16;
    const int hi = lane >> 4;
    const int n_ = lane & 15;
    const int n  = d0 + n_;

    const __bf16* Abase = attn + b * NHEAD * SEQ * SEQ + s * SEQ;  // m stride S*T
    v8f acc = {};
    for (int t0 = 0; t0 < SEQ; t0 += 64) {
        v16bf aA = load_a_frag(Abase, SEQ * SEQ, 0, t0,      lane);
        v16bf aB = load_a_frag(Abase, SEQ * SEQ, 0, t0 + 32, lane);
        v16bf bbA, bbB;
#pragma unroll
        for (int e = 0; e < 16; ++e) {
            const int ta = t0 + hi * 16 + e;
            bbA[e] = relv[clamp_rel(s - ta) * DK + n];
            bbB[e] = relv[clamp_rel(s - (ta + 32)) * DK + n];
        }
        acc = __builtin_amdgcn_wmma_f32_16x16x32_bf16(false, aA, false, bbA, (short)0, acc, false, false);
        acc = __builtin_amdgcn_wmma_f32_16x16x32_bf16(false, aB, false, bbB, (short)0, acc, false, false);
    }
#pragma unroll
    for (int v = 0; v < 8; ++v) {
        const int hrow = v + 8 * hi;
        out[(s * BATCH + b) * DMODEL + hrow * DK + d0 + n_] += acc[v];
    }
}

// ---------------------------------------------------------------------------
extern "C" void kernel_launch(void* const* d_in, const int* in_sizes, int n_in,
                              void* d_out, int out_size, void* d_ws, size_t ws_size,
                              hipStream_t stream) {
    (void)in_sizes; (void)n_in; (void)out_size; (void)ws_size;
    const float* x    = (const float*)d_in[0];
    const float* w_q  = (const float*)d_in[1];
    const float* w_k  = (const float*)d_in[2];
    const float* w_v  = (const float*)d_in[3];
    const float* w_o  = (const float*)d_in[4];
    const float* b_o  = (const float*)d_in[5];
    const float* relk = (const float*)d_in[6];
    const float* relv = (const float*)d_in[7];
    float* out = (float*)d_out;

    // ---- workspace partition (all offsets 256B aligned) ----
    char* p = (char*)d_ws;
    auto take = [&](size_t bytes) { char* r = p; p += (bytes + 255) & ~size_t(255); return r; };
    __bf16* x_bf    = (__bf16*)take((size_t)NTOK * DMODEL * 2);
    __bf16* wq_bf   = (__bf16*)take((size_t)DMODEL * DMODEL * 2);
    __bf16* wk_bf   = (__bf16*)take((size_t)DMODEL * DMODEL * 2);
    __bf16* wv_bf   = (__bf16*)take((size_t)DMODEL * DMODEL * 2);
    __bf16* wo_bf   = (__bf16*)take((size_t)DMODEL * DMODEL * 2);
    __bf16* relk_bf = (__bf16*)take((size_t)RELBUF * DK * 2);
    __bf16* relv_bf = (__bf16*)take((size_t)RELBUF * DK * 2);
    __bf16* Q_bf    = (__bf16*)take((size_t)NTOK * DMODEL * 2);
    __bf16* K_bf    = (__bf16*)take((size_t)NTOK * DMODEL * 2);
    __bf16* V_bf    = (__bf16*)take((size_t)NTOK * DMODEL * 2);
    __bf16* Vt_bf   = (__bf16*)take((size_t)NTOK * DMODEL * 2);
    float*  Pbuf    = (float*) take((size_t)NTOK * NHEAD * RELPAD * 4);
    __bf16* attn    = (__bf16*)take((size_t)BATCH * NHEAD * SEQ * SEQ * 2);
    float*  ao      = (float*) take((size_t)NTOK * DMODEL * 4);
    __bf16* ao_bf   = (__bf16*)take((size_t)NTOK * DMODEL * 2);

    auto cast = [&](const float* s, __bf16* d, int nsrc, int ntot) {
        cast_bf16_kernel<<<(ntot + 255) / 256, 256, 0, stream>>>(s, d, nsrc, ntot);
    };
    cast(x,   x_bf,  NTOK * DMODEL,  NTOK * DMODEL);
    cast(w_q, wq_bf, DMODEL * DMODEL, DMODEL * DMODEL);
    cast(w_k, wk_bf, DMODEL * DMODEL, DMODEL * DMODEL);
    cast(w_v, wv_bf, DMODEL * DMODEL, DMODEL * DMODEL);
    cast(w_o, wo_bf, DMODEL * DMODEL, DMODEL * DMODEL);
    cast(relk, relk_bf, RELROWS * DK, RELBUF * DK);
    cast(relv, relv_bf, RELROWS * DK, RELBUF * DK);

    // Q/K/V projections: [4096,1024] = x_bf @ W^T ; bf16 output.
    dim3 gproj(NTOK / 128, DMODEL / 64, 1);
    gemm_bf16_kernel<<<gproj, 256, 0, stream>>>(x_bf, 0, DMODEL, wq_bf, 0, DMODEL,
        nullptr, Q_bf, 0, DMODEL, DMODEL, DMODEL, nullptr);
    gemm_bf16_kernel<<<gproj, 256, 0, stream>>>(x_bf, 0, DMODEL, wk_bf, 0, DMODEL,
        nullptr, K_bf, 0, DMODEL, DMODEL, DMODEL, nullptr);
    gemm_bf16_kernel<<<gproj, 256, 0, stream>>>(x_bf, 0, DMODEL, wv_bf, 0, DMODEL,
        nullptr, V_bf, 0, DMODEL, DMODEL, DMODEL, nullptr);

    // Transpose V -> Vt[(b*1024 + h*64 + d), s]
    transpose_v_kernel<<<dim3(SEQ / 32, DMODEL / 32, BATCH), dim3(32, 8), 0, stream>>>(
        V_bf, Vt_bf);

    // P[(s,b,h), r] = Q @ rel_k^T : M=65536, N=272 (stores guarded; B reads
    // from 320-row zero-padded table), K=64, f32 out.
    dim3 gp((NTOK * NHEAD) / 128, RELBUF / 64, 1);
    gemm_bf16_kernel<<<gp, 256, 0, stream>>>(Q_bf, 0, DK, relk_bf, 0, DK,
        Pbuf, nullptr, 0, RELPAD, RELPAD, DK, nullptr);

    // fused scores + softmax over heads -> attn (bf16, [b,h,s,t])
    scores_softmax_kernel<<<dim3(SEQ / 16, SEQ / 16, BATCH), 512, 0, stream>>>(
        Q_bf, K_bf, Pbuf, attn);

    // out1[s,b,h,d] = sum_t attn[b,h,s,t] * Vt[(b,h,d), t] : batched z=b*16+h.
    gemm_bf16_kernel<<<dim3(SEQ / 128, 1, BATCH * NHEAD), 256, 0, stream>>>(
        attn, (long)SEQ * SEQ, SEQ,
        Vt_bf, (long)DK * SEQ, SEQ,
        ao, nullptr, DK, BATCH * DMODEL,
        DK, SEQ, nullptr);

    // out2 += attn @ gathered rel_v
    attn_relv_kernel<<<dim3(SEQ, BATCH, 1), 128, 0, stream>>>(attn, relv_bf, ao);

    // final projection: out = ao_bf @ w_o^T + b_o (f32 out)
    cast(ao, ao_bf, NTOK * DMODEL, NTOK * DMODEL);
    gemm_bf16_kernel<<<gproj, 256, 0, stream>>>(ao_bf, 0, DMODEL, wo_bf, 0, DMODEL,
        out, nullptr, 0, DMODEL, DMODEL, DMODEL, b_o);
}
